// MultimodalPhiModel_63350767616417
// MI455X (gfx1250) — matrix operand order repaired and checked by reading, not compiled, not run on hardware
//
#include <hip/hip_runtime.h>

// ---------------- model constants ----------------
#define NLAYER 8
#define DMODEL 2560
#define NHEAD 32
#define HDIM 80          // DMODEL / NHEAD
#define ROTD 32
#define FFDIM 10240      // 4*D
#define NVOCAB 51200
#define CLIPD 768
#define BATCH 4
#define TSTEPS 16
#define S0 92            // 3 + 49 + 4 + 32 + 4
#define SMAXP 112        // padded per-batch capacity (4*112 = 448 = 14*32)
#define RMAX (BATCH*SMAXP)
#define IMGROWS 196      // 4*49
#define IMGROWSP 224     // padded to mult of 32
#define LMROWS 32        // padded last-token rows (mult of 32)

// input pointer order: setup_inputs() dict order, 'layers' expanded in its
// own dict insertion order.
enum {
  IN_IMG = 0, IN_HUMAN, IN_GPT, IN_EMBED, IN_WIMG,
  IN_RB_LNG, IN_RB_LNB, IN_RB_W1, IN_RB_B1, IN_RB_W2, IN_RB_B2,
  IN_L_LNG, IN_L_LNB, IN_L_WQ, IN_L_BQ, IN_L_WK, IN_L_BK, IN_L_WV, IN_L_BV,
  IN_L_WO, IN_L_BO, IN_L_W1, IN_L_B1, IN_L_W2, IN_L_B2,
  IN_FLNG, IN_FLNB, IN_LMW, IN_LMB, IN_FIXED
};

// ---------------- vector types for WMMA ----------------
typedef __attribute__((ext_vector_type(16))) __bf16 v16bf;
typedef __attribute__((ext_vector_type(8)))  float  v8f;
typedef __attribute__((ext_vector_type(4)))  unsigned int v4u;
typedef __attribute__((ext_vector_type(8)))  unsigned int v8u;

union AFrag { v4u q[2]; v16bf bf; };
union BFrag { v8u q;    v16bf bf; };

__device__ __forceinline__ unsigned short f2bf_bits(float f) {
  unsigned u = __float_as_uint(f);
  return (unsigned short)((u + 0x7FFFu + ((u >> 16) & 1u)) >> 16);   // RNE
}

__device__ __forceinline__ float gelu_tanh(float x) {
  float x3 = x * x * x;
  float t = 0.7978845608028654f * (x + 0.044715f * x3);
  return 0.5f * x * (1.0f + tanhf(t));
}

// ---------------- weight fp32 [K,N] -> bf16 transposed [N,K] ----------------
__global__ __launch_bounds__(256)
void k_transpose_bf16(const float* __restrict__ src, unsigned short* __restrict__ dst,
                      int K, int N) {
  __shared__ float t[32][33];
  int n0 = blockIdx.x * 32, k0 = blockIdx.y * 32;
  int tx = threadIdx.x & 31, ty = threadIdx.x >> 5;   // 32 x 8
#pragma unroll
  for (int i = 0; i < 32; i += 8)
    t[ty + i][tx] = src[(size_t)(k0 + ty + i) * N + n0 + tx];
  __syncthreads();
#pragma unroll
  for (int i = 0; i < 32; i += 8)
    dst[(size_t)(n0 + ty + i) * K + k0 + tx] = f2bf_bits(t[tx][ty + i]);
}

// ---------------- f32 -> bf16 row copy ----------------
__global__ void k_f32_to_bf16(const float* __restrict__ s, unsigned short* __restrict__ d, int n) {
  int i = blockIdx.x * blockDim.x + threadIdx.x;
  if (i < n) d[i] = f2bf_bits(s[i]);
}

// ---------------- WMMA bf16 GEMM: C[M,N] = act(A[M,K] @ BT[N,K]^T + bias) ----
// Each wave computes a 32x64 tile of C (2 A-frags x 4 B-frags = 8 WMMAs/K-step);
// 4 waves (128 thr) per block. Requires M%32==0, N%64==0, K%32==0.
__global__ __launch_bounds__(128)
void k_gemm_bf16(const unsigned short* __restrict__ A, const unsigned short* __restrict__ BT,
                 const float* __restrict__ bias, float* __restrict__ C,
                 unsigned short* __restrict__ Cb, int M, int N, int K, int Mvalid, int act) {
  int wave = blockIdx.x * (blockDim.x >> 5) + (threadIdx.x >> 5);
  int MT = M >> 5, NG = N >> 6;
  if (wave >= MT * NG) return;                 // wave-uniform exit, EXEC stays full
  int mt = wave % MT, ng = wave / MT;
  int lane = threadIdx.x & 31;
  int hsel = lane >> 4;                        // lane-group select
  int l15  = lane & 15;

  const unsigned short* arow0 = A  + (size_t)(mt * 32 + l15) * K;
  const unsigned short* arow1 = A  + (size_t)(mt * 32 + 16 + l15) * K;
  const unsigned short* bcol0 = BT + (size_t)(ng * 64 + l15) * K;

  v8f acc[8] = {};
  for (int kb = 0; kb < K; kb += 32) {
    AFrag a0, a1;
    // ISA 7.12.2 16-bit A layout: lanes 0-15 K{kb..kb+7, kb+16..kb+23}, lanes 16-31 +8
    a0.q[0] = *(const v4u*)(arow0 + kb + hsel * 8);
    a0.q[1] = *(const v4u*)(arow0 + kb + 16 + hsel * 8);
    a1.q[0] = *(const v4u*)(arow1 + kb + hsel * 8);
    a1.q[1] = *(const v4u*)(arow1 + kb + 16 + hsel * 8);
#pragma unroll
    for (int j = 0; j < 4; ++j) {
      BFrag b;
      // B layout: same K order as A: K{kb..kb+7, kb+16..kb+23}, lanes 16-31 +8
      b.q = __builtin_shufflevector(*(const v4u*)(bcol0 + (size_t)(j * 16) * K + kb + hsel * 8), *(const v4u*)(bcol0 + (size_t)(j * 16) * K + kb + 16 + hsel * 8), 0, 1, 2, 3, 4, 5, 6, 7);
      acc[j]     = __builtin_amdgcn_wmma_f32_16x16x32_bf16(
          false, a0.bf, false, b.bf, (short)0, acc[j],     false, false);
      acc[4 + j] = __builtin_amdgcn_wmma_f32_16x16x32_bf16(
          false, a1.bf, false, b.bf, (short)0, acc[4 + j], false, false);
    }
  }
#pragma unroll
  for (int g = 0; g < 2; ++g) {
#pragma unroll
    for (int j = 0; j < 4; ++j) {
      int col = ng * 64 + j * 16 + l15;
      float bv = bias ? bias[col] : 0.0f;
#pragma unroll
      for (int r = 0; r < 8; ++r) {
        int row = mt * 32 + g * 16 + r + hsel * 8;   // C layout: VGPR r -> M=r / r+8
        if (row < Mvalid) {
          float v = acc[g * 4 + j][r] + bv;
          if (act == 1) v = gelu_tanh(v);
          C[(size_t)row * N + col] = v;
          if (Cb) Cb[(size_t)row * N + col] = f2bf_bits(v);
        }
      }
    }
  }
}

// ---------------- LayerNorm (one block per row) ----------------
__global__ __launch_bounds__(256)
void k_layernorm(const float* __restrict__ x, const float* __restrict__ g,
                 const float* __restrict__ be, float* __restrict__ h,
                 unsigned short* __restrict__ hb, int D) {
  __shared__ float r1[256], r2[256];
  int row = blockIdx.x, tid = threadIdx.x;
  const float* xr = x + (size_t)row * D;
  float s = 0.f, ss = 0.f;
  for (int i = tid; i < D; i += 256) { float v = xr[i]; s += v; ss += v * v; }
  r1[tid] = s; r2[tid] = ss; __syncthreads();
  for (int st = 128; st > 0; st >>= 1) {
    if (tid < st) { r1[tid] += r1[tid + st]; r2[tid] += r2[tid + st]; }
    __syncthreads();
  }
  float mu  = r1[0] / D;
  float var = r2[0] / D - mu * mu;
  float rr  = rsqrtf(var + 1e-5f);
  for (int i = tid; i < D; i += 256) {
    float v = (xr[i] - mu) * rr * g[i] + be[i];
    h[(size_t)row * D + i] = v;
    if (hb) hb[(size_t)row * D + i] = f2bf_bits(v);
  }
}

// ---------------- RoPE on q,k (first 32 dims of each head) ----------------
__global__ void k_rope(float* __restrict__ q, float* __restrict__ k, int S) {
  int idx = blockIdx.x * blockDim.x + threadIdx.x;
  int total = BATCH * S * NHEAD * 16;
  if (idx >= total) return;
  int i  = idx & 15;
  int hh = (idx >> 4) % NHEAD;
  int rs = (idx >> 4) / NHEAD;
  int s  = rs % S, b = rs / S;
  float inv = __powf(10000.0f, -((float)i) / 16.0f);
  float ang = (float)s * inv;
  float c = __cosf(ang), sn = __sinf(ang);
  size_t base = (size_t)(b * SMAXP + s) * DMODEL + hh * HDIM;
  float x1 = q[base + i], x2 = q[base + 16 + i];
  q[base + i]      = x1 * c - x2 * sn;
  q[base + 16 + i] = x2 * c + x1 * sn;
  x1 = k[base + i]; x2 = k[base + 16 + i];
  k[base + i]      = x1 * c - x2 * sn;
  k[base + 16 + i] = x2 * c + x1 * sn;
}

// ---------------- causal attention, online softmax per (b,h,q) ----------------
__global__ __launch_bounds__(128)
void k_attention(const float* __restrict__ q, const float* __restrict__ k,
                 const float* __restrict__ v, float* __restrict__ o, int S) {
  int idx = blockIdx.x * blockDim.x + threadIdx.x;
  int total = BATCH * NHEAD * S;
  if (idx >= total) return;
  int qi = idx % S; int t = idx / S; int hh = t % NHEAD; int b = t / NHEAD;
  const float scale = 0.11180339887498949f;    // 1/sqrt(80)
  size_t rowq = (size_t)(b * SMAXP + qi) * DMODEL + hh * HDIM;
  float qr[HDIM];
#pragma unroll
  for (int d = 0; d < HDIM; ++d) qr[d] = q[rowq + d];
  float m = -3.4e38f, l = 0.f, av[HDIM];
#pragma unroll
  for (int d = 0; d < HDIM; ++d) av[d] = 0.f;
  for (int kj = 0; kj <= qi; ++kj) {
    size_t rk = (size_t)(b * SMAXP + kj) * DMODEL + hh * HDIM;
    float s = 0.f;
#pragma unroll
    for (int d = 0; d < HDIM; ++d) s += qr[d] * k[rk + d];
    s *= scale;
    float mn = fmaxf(m, s);
    float corr = __expf(m - mn), p = __expf(s - mn);
    l = l * corr + p;
#pragma unroll
    for (int d = 0; d < HDIM; ++d) av[d] = av[d] * corr + p * v[rk + d];
    m = mn;
  }
  float inv = 1.f / l;
#pragma unroll
  for (int d = 0; d < HDIM; ++d) o[rowq + d] = av[d] * inv;
}

// ---------------- misc elementwise ----------------
__global__ void k_add2(const float* a, const float* b, float* c, int n) {
  int i = blockIdx.x * blockDim.x + threadIdx.x;
  if (i < n) c[i] = a[i] + b[i];
}
__global__ void k_resid(float* x, const float* a, const float* b, int n) {
  int i = blockIdx.x * blockDim.x + threadIdx.x;
  if (i < n) x[i] += a[i] + b[i];
}

// ---------------- build the initial 92-token sequence ----------------
__global__ void k_build_inputs(const float* __restrict__ embed, const int* __restrict__ fixed_ids,
                               const int* __restrict__ human_ids, const float* __restrict__ ip,
                               float* __restrict__ x) {
  int idx = blockIdx.x * blockDim.x + threadIdx.x;
  int total = BATCH * S0 * DMODEL;
  if (idx >= total) return;
  int d = idx % DMODEL;
  int s = (idx / DMODEL) % S0;
  int b = idx / (DMODEL * S0);
  float val;
  if (s < 3)       val = embed[(size_t)fixed_ids[s] * DMODEL + d];
  else if (s < 52) val = ip[(size_t)(b * 49 + (s - 3)) * DMODEL + d];
  else if (s < 56) val = embed[(size_t)fixed_ids[3 + (s - 52)] * DMODEL + d];
  else if (s < 88) val = embed[(size_t)human_ids[b * 32 + (s - 56)] * DMODEL + d];
  else             val = embed[(size_t)fixed_ids[7 + (s - 88)] * DMODEL + d];
  x[(size_t)(b * SMAXP + s) * DMODEL + d] = val;
}

// ---------------- gather last tokens (rows BATCH..LMROWS-1 zeroed) ----------
__global__ void k_gather_last(const float* __restrict__ x, float* __restrict__ last, int S) {
  int idx = blockIdx.x * blockDim.x + threadIdx.x;
  if (idx >= LMROWS * DMODEL) return;
  int d = idx % DMODEL, r = idx / DMODEL;
  float v = 0.f;
  if (r < BATCH) v = x[(size_t)(r * SMAXP + S - 1) * DMODEL + d];
  last[idx] = v;
}

// ---------------- argmax over vocab (first-max tie break, like jnp) ----------
__global__ __launch_bounds__(256)
void k_argmax(const float* __restrict__ logits, int* __restrict__ tok) {
  __shared__ float bv[256]; __shared__ int bi[256];
  int b = blockIdx.x, tid = threadIdx.x;
  const float* lr = logits + (size_t)b * NVOCAB;
  float best = -3.4e38f; int bidx = 0x7fffffff;
  for (int i = tid; i < NVOCAB; i += 256) {
    float v = lr[i];
    if (v > best || (v == best && i < bidx)) { best = v; bidx = i; }
  }
  bv[tid] = best; bi[tid] = bidx; __syncthreads();
  for (int st = 128; st > 0; st >>= 1) {
    if (tid < st) {
      if (bv[tid + st] > bv[tid] || (bv[tid + st] == bv[tid] && bi[tid + st] < bi[tid])) {
        bv[tid] = bv[tid + st]; bi[tid] = bi[tid + st];
      }
    }
    __syncthreads();
  }
  if (tid == 0) tok[b] = bi[0];
}

// ---------------- append token embedding + loss term ----------------
__global__ __launch_bounds__(256)
void k_finish(const float* __restrict__ embed, const int* __restrict__ tok,
              const int* __restrict__ gpt_ids, float* __restrict__ x,
              float* __restrict__ loss_part, float* __restrict__ out_tok,
              int S, int step) {
  __shared__ float red[256];
  int b = blockIdx.x, tid = threadIdx.x;
  int t = tok[b];
  const float* e = embed + (size_t)t * DMODEL;
  float* xr = x + (size_t)(b * SMAXP + S) * DMODEL;
  for (int i = tid; i < DMODEL; i += 256) xr[i] = e[i];
  // logsumexp(e)
  float mx = -3.4e38f;
  for (int i = tid; i < DMODEL; i += 256) mx = fmaxf(mx, e[i]);
  red[tid] = mx; __syncthreads();
  for (int st = 128; st > 0; st >>= 1) { if (tid < st) red[tid] = fmaxf(red[tid], red[tid + st]); __syncthreads(); }
  mx = red[0]; __syncthreads();
  float se = 0.f;
  for (int i = tid; i < DMODEL; i += 256) se += __expf(e[i] - mx);
  red[tid] = se; __syncthreads();
  for (int st = 128; st > 0; st >>= 1) { if (tid < st) red[tid] += red[tid + st]; __syncthreads(); }
  float lse = mx + __logf(red[0]); __syncthreads();
  // dot with gpt-target embedding
  int gid = gpt_ids[b * TSTEPS + step];
  const float* ge = embed + (size_t)gid * DMODEL;
  float s1 = 0.f, s2 = 0.f;
  for (int i = tid; i < DMODEL; i += 256) { float gv = ge[i]; s1 += gv * e[i]; s2 += gv; }
  red[tid] = s1; __syncthreads();
  for (int st = 128; st > 0; st >>= 1) { if (tid < st) red[tid] += red[tid + st]; __syncthreads(); }
  s1 = red[0]; __syncthreads();
  red[tid] = s2; __syncthreads();
  for (int st = 128; st > 0; st >>= 1) { if (tid < st) red[tid] += red[tid + st]; __syncthreads(); }
  s2 = red[0];
  if (tid == 0) {
    loss_part[step * BATCH + b] = -(s1 - s2 * lse) * (1.0f / BATCH);
    out_tok[b * TSTEPS + step] = (float)t;
  }
}

__global__ void k_finalize(const float* __restrict__ loss_part, float* __restrict__ out) {
  if (blockIdx.x == 0 && threadIdx.x == 0) {
    float s = 0.f;
    for (int i = 0; i < TSTEPS * BATCH; ++i) s += loss_part[i];  // fixed order => deterministic
    out[0] = s / (float)(TSTEPS - 1);
  }
}

// =====================================================================
extern "C" void kernel_launch(void* const* d_in, const int* in_sizes, int n_in,
                              void* d_out, int out_size, void* d_ws, size_t ws_size,
                              hipStream_t stream) {
  (void)in_sizes; (void)n_in; (void)out_size; (void)ws_size;
  const float* f_img   = (const float*)d_in[IN_IMG];
  const int*   human   = (const int*)d_in[IN_HUMAN];
  const int*   gpt     = (const int*)d_in[IN_GPT];
  const float* embed   = (const float*)d_in[IN_EMBED];
  const float* w_img   = (const float*)d_in[IN_WIMG];
  const float* rb_lng  = (const float*)d_in[IN_RB_LNG];
  const float* rb_lnb  = (const float*)d_in[IN_RB_LNB];
  const float* rb_w1   = (const float*)d_in[IN_RB_W1];
  const float* rb_b1   = (const float*)d_in[IN_RB_B1];
  const float* rb_w2   = (const float*)d_in[IN_RB_W2];
  const float* rb_b2   = (const float*)d_in[IN_RB_B2];
  const float* l_lng   = (const float*)d_in[IN_L_LNG];
  const float* l_lnb   = (const float*)d_in[IN_L_LNB];
  const float* l_wq    = (const float*)d_in[IN_L_WQ];
  const float* l_bq    = (const float*)d_in[IN_L_BQ];
  const float* l_wk    = (const float*)d_in[IN_L_WK];
  const float* l_bk    = (const float*)d_in[IN_L_BK];
  const float* l_wv    = (const float*)d_in[IN_L_WV];
  const float* l_bv    = (const float*)d_in[IN_L_BV];
  const float* l_wo    = (const float*)d_in[IN_L_WO];
  const float* l_bo    = (const float*)d_in[IN_L_BO];
  const float* l_w1    = (const float*)d_in[IN_L_W1];
  const float* l_b1    = (const float*)d_in[IN_L_B1];
  const float* l_w2    = (const float*)d_in[IN_L_W2];
  const float* l_b2    = (const float*)d_in[IN_L_B2];
  const float* flng    = (const float*)d_in[IN_FLNG];
  const float* flnb    = (const float*)d_in[IN_FLNB];
  const float* lmw     = (const float*)d_in[IN_LMW];
  const float* lmb     = (const float*)d_in[IN_LMB];
  const int*   fixed   = (const int*)d_in[IN_FIXED];
  float* out = (float*)d_out;

  // ---------- workspace layout (256B aligned offsets) ----------
  char* ws = (char*)d_ws;
  size_t off = 0;
  auto alloc = [&](size_t bytes) { size_t o = off; off += (bytes + 255) & ~(size_t)255; return o; };
  const size_t DD = (size_t)DMODEL * DMODEL, DF = (size_t)DMODEL * FFDIM;

  float* lossp = (float*)(ws + alloc(TSTEPS * BATCH * 4));
  int*   tok   = (int*)(ws + alloc(BATCH * 4));
  unsigned short* wimgT = (unsigned short*)(ws + alloc((size_t)CLIPD * DMODEL * 2));
  unsigned short* rbw1T = (unsigned short*)(ws + alloc(DD * 2));
  unsigned short* rbw2T = (unsigned short*)(ws + alloc(DD * 2));
  unsigned short* wqT = (unsigned short*)(ws + alloc(NLAYER * DD * 2));
  unsigned short* wkT = (unsigned short*)(ws + alloc(NLAYER * DD * 2));
  unsigned short* wvT = (unsigned short*)(ws + alloc(NLAYER * DD * 2));
  unsigned short* woT = (unsigned short*)(ws + alloc(NLAYER * DD * 2));
  unsigned short* w1T = (unsigned short*)(ws + alloc(NLAYER * DF * 2));
  unsigned short* w2T = (unsigned short*)(ws + alloc(NLAYER * DF * 2));
  unsigned short* lmT = (unsigned short*)(ws + alloc((size_t)DMODEL * NVOCAB * 2));
  unsigned short* imgb  = (unsigned short*)(ws + alloc((size_t)IMGROWSP * CLIPD * 2));
  float* ip0  = (float*)(ws + alloc((size_t)IMGROWSP * DMODEL * 4));
  float* iph  = (float*)(ws + alloc((size_t)IMGROWSP * DMODEL * 4));
  unsigned short* iphb = (unsigned short*)(ws + alloc((size_t)IMGROWSP * DMODEL * 2));
  float* ipt  = (float*)(ws + alloc((size_t)IMGROWSP * DMODEL * 4));
  unsigned short* iptb = (unsigned short*)(ws + alloc((size_t)IMGROWSP * DMODEL * 2));
  float* ipu  = (float*)(ws + alloc((size_t)IMGROWSP * DMODEL * 4));
  float* x    = (float*)(ws + alloc((size_t)RMAX * DMODEL * 4));
  float* h    = (float*)(ws + alloc((size_t)RMAX * DMODEL * 4));
  unsigned short* hb = (unsigned short*)(ws + alloc((size_t)RMAX * DMODEL * 2));
  float* qb_  = (float*)(ws + alloc((size_t)RMAX * DMODEL * 4));
  float* kb_  = (float*)(ws + alloc((size_t)RMAX * DMODEL * 4));
  float* vb_  = (float*)(ws + alloc((size_t)RMAX * DMODEL * 4));
  float* ob_  = (float*)(ws + alloc((size_t)RMAX * DMODEL * 4));
  unsigned short* obb = (unsigned short*)(ws + alloc((size_t)RMAX * DMODEL * 2));
  float* ao   = (float*)(ws + alloc((size_t)RMAX * DMODEL * 4));
  float* m1   = (float*)(ws + alloc((size_t)RMAX * FFDIM * 4));
  unsigned short* m1b = (unsigned short*)(ws + alloc((size_t)RMAX * FFDIM * 2));
  float* mo   = (float*)(ws + alloc((size_t)RMAX * DMODEL * 4));
  float* last = (float*)(ws + alloc((size_t)LMROWS * DMODEL * 4));
  float* lasth = (float*)(ws + alloc((size_t)LMROWS * DMODEL * 4));
  unsigned short* lasthb = (unsigned short*)(ws + alloc((size_t)LMROWS * DMODEL * 2));
  float* logits = (float*)(ws + alloc((size_t)LMROWS * NVOCAB * 4));

  auto xpose = [&](const float* src, unsigned short* dst, int K, int N) {
    dim3 g(N / 32, K / 32);
    k_transpose_bf16<<<g, 256, 0, stream>>>(src, dst, K, N);
  };
  auto gemm = [&](const unsigned short* A, const unsigned short* BT, const float* bias,
                  float* C, unsigned short* Cb, int M, int N, int K, int Mvalid, int act) {
    int waves = (M / 32) * (N / 64);
    k_gemm_bf16<<<(waves + 3) / 4, 128, 0, stream>>>(A, BT, bias, C, Cb, M, N, K, Mvalid, act);
  };

  // ---------- one-time weight convert+transpose (bf16, [N,K]) ----------
  xpose(w_img, wimgT, CLIPD, DMODEL);
  xpose(rb_w1, rbw1T, DMODEL, DMODEL);
  xpose(rb_w2, rbw2T, DMODEL, DMODEL);
  for (int l = 0; l < NLAYER; ++l) {
    xpose(l_wq + l * DD, wqT + l * DD, DMODEL, DMODEL);
    xpose(l_wk + l * DD, wkT + l * DD, DMODEL, DMODEL);
    xpose(l_wv + l * DD, wvT + l * DD, DMODEL, DMODEL);
    xpose(l_wo + l * DD, woT + l * DD, DMODEL, DMODEL);
    xpose(l_w1 + l * DF, w1T + l * DF, DMODEL, FFDIM);
    xpose(l_w2 + l * DF, w2T + l * DF, FFDIM, DMODEL);
  }
  xpose(lmw, lmT, DMODEL, NVOCAB);

  // ---------- image projection resblock ----------
  {
    int n = IMGROWS * CLIPD;
    k_f32_to_bf16<<<(n + 255) / 256, 256, 0, stream>>>(f_img, imgb, n);
    gemm(imgb, wimgT, nullptr, ip0, nullptr, IMGROWSP, DMODEL, CLIPD, IMGROWS, 0);
    k_layernorm<<<IMGROWS, 256, 0, stream>>>(ip0, rb_lng, rb_lnb, iph, iphb, DMODEL);
    gemm(iphb, rbw1T, rb_b1, ipt, iptb, IMGROWSP, DMODEL, DMODEL, IMGROWS, 1);
    gemm(iptb, rbw2T, rb_b2, ipu, nullptr, IMGROWSP, DMODEL, DMODEL, IMGROWS, 0);
    int m = IMGROWS * DMODEL;
    k_add2<<<(m + 255) / 256, 256, 0, stream>>>(iph, ipu, ip0, m);  // ip = h + gelu(..)@w2 + b2
  }

  // ---------- initial token/image sequence ----------
  {
    int n = BATCH * S0 * DMODEL;
    k_build_inputs<<<(n + 255) / 256, 256, 0, stream>>>(embed, fixed, human, ip0, x);
  }

  // ---------- autoregressive decode ----------
  for (int step = 0; step < TSTEPS; ++step) {
    int S = S0 + step;
    for (int l = 0; l < NLAYER; ++l) {
      k_layernorm<<<RMAX, 256, 0, stream>>>(x, l_lng + l * DMODEL, l_lnb + l * DMODEL, h, hb, DMODEL);
      gemm(hb, wqT + l * DD, l_bq + l * DMODEL, qb_, nullptr, RMAX, DMODEL, DMODEL, RMAX, 0);
      gemm(hb, wkT + l * DD, l_bk + l * DMODEL, kb_, nullptr, RMAX, DMODEL, DMODEL, RMAX, 0);
      gemm(hb, wvT + l * DD, l_bv + l * DMODEL, vb_, nullptr, RMAX, DMODEL, DMODEL, RMAX, 0);
      int nr = BATCH * S * NHEAD * 16;
      k_rope<<<(nr + 255) / 256, 256, 0, stream>>>(qb_, kb_, S);
      int na = BATCH * NHEAD * S;
      k_attention<<<(na + 127) / 128, 128, 0, stream>>>(qb_, kb_, vb_, ob_, S);
      int ne = RMAX * DMODEL;
      k_f32_to_bf16<<<(ne + 255) / 256, 256, 0, stream>>>(ob_, obb, ne);
      gemm(obb, woT + l * DD, l_bo + l * DMODEL, ao, nullptr, RMAX, DMODEL, DMODEL, RMAX, 0);
      gemm(hb, w1T + l * DF, l_b1 + l * FFDIM, m1, m1b, RMAX, FFDIM, DMODEL, RMAX, 1);
      gemm(m1b, w2T + l * DF, l_b2 + l * DMODEL, mo, nullptr, RMAX, DMODEL, FFDIM, RMAX, 0);
      k_resid<<<(ne + 255) / 256, 256, 0, stream>>>(x, ao, mo, ne);
    }
    // final LN + lm_head on last token of each batch only
    int ng = LMROWS * DMODEL;
    k_gather_last<<<(ng + 255) / 256, 256, 0, stream>>>(x, last, S);
    k_layernorm<<<LMROWS, 256, 0, stream>>>(last, flng, flnb, lasth, lasthb, DMODEL);
    gemm(lasthb, lmT, lmb, logits, nullptr, LMROWS, NVOCAB, DMODEL, LMROWS, 0);
    k_argmax<<<BATCH, 256, 0, stream>>>(logits, tok);
    k_finish<<<BATCH, 256, 0, stream>>>(embed, tok, gpt, x, lossp, out + 1, S, step);
  }
  k_finalize<<<1, 1, 0, stream>>>(lossp, out);
}
